// MultiHeadedLatentAttention_30227979829383
// MI455X (gfx1250) — compile-verified, matrix operand order checked by reading, not run on hardware
//
#include <hip/hip_runtime.h>
#include <cstdint>

// ---------------------------------------------------------------------------
// MLA forward for MI455X (gfx1250, wave32, WMMA bf16, async global->LDS).
//   B=4 S=1024 H=4096 NH=32 QL=1536 KVL=512 RD=64 ND=128 VD=128 QHD=192
// ---------------------------------------------------------------------------

typedef unsigned short u16;
typedef unsigned int   u32;
typedef unsigned long long u64;
typedef __attribute__((ext_vector_type(16))) __bf16 v16bf;
typedef __attribute__((ext_vector_type(8)))  float  v8f;

#define B_   4
#define S_   1024
#define H_   4096
#define NH_  32
#define QL_  1536
#define KVL_ 512
#define RD_  64
#define ND_  128
#define VD_  128
#define QHD_ 192
#define ROWS_ (B_ * S_)            // 4096

// fp32 -> bf16 round-to-nearest-even
__device__ __forceinline__ u16 f2bf(float f) {
  u32 x = __float_as_uint(f);
  u32 r = x + 0x7FFFu + ((x >> 16) & 1u);
  return (u16)(r >> 16);
}

// Low 32 bits of a generic LDS pointer are the workgroup-relative LDS byte
// address.
__device__ __forceinline__ u32 lds_off(const void* p) {
  return (u32)(uintptr_t)p;
}

// CDNA5 async global->LDS copy, 16B per lane, tracked by ASYNCcnt.
__device__ __forceinline__ void async_b128(u32 loff, const void* base, u32 goff) {
  asm volatile("global_load_async_to_lds_b128 %0, %1, %2"
               :: "v"(loff), "v"(goff), "s"((u64)(uintptr_t)base)
               : "memory");
}
__device__ __forceinline__ void async_wait0() {
  asm volatile("s_wait_asynccnt 0x0" ::: "memory");
}

// Load a 16x32 bf16 WMMA fragment from an LDS row base.
// Lanes 0-15 hold K={0..7,16..23}, lanes 16-31 hold K={8..15,24..31};
// each half is one ds_load_b128.
__device__ __forceinline__ v16bf ldfrag(const u16* rb, int half) {
  union { v16bf v; uint4 q[2]; } u;
  u.q[0] = *(const uint4*)(rb + half * 8);
  u.q[1] = *(const uint4*)(rb + 16 + half * 8);
  return u.v;
}

__device__ __forceinline__ v8f wmma_bf16(v16bf a, v16bf b, v8f c) {
  return __builtin_amdgcn_wmma_f32_16x16x32_bf16(false, a, false, b,
                                                 (short)0, c, false, false);
}

// ---------------------------------------------------------------------------
// Elementwise casts
// ---------------------------------------------------------------------------
__global__ __launch_bounds__(256) void cast_bf16_kernel(
    const float* __restrict__ X, u16* __restrict__ Y, int n4) {
  int idx = blockIdx.x * 256 + threadIdx.x;
  if (idx >= n4) return;
  float4 v = ((const float4*)X)[idx];
  u32 lo = (u32)f2bf(v.x) | ((u32)f2bf(v.y) << 16);
  u32 hi = (u32)f2bf(v.z) | ((u32)f2bf(v.w) << 16);
  ((uint2*)Y)[idx] = make_uint2(lo, hi);
}

// W [K,N] fp32 -> Wt [Npad,K] bf16 (rows n>=N zero-filled)
__global__ __launch_bounds__(256) void transpose_cast_kernel(
    const float* __restrict__ W, u16* __restrict__ Wt, int K, int N, int Npad) {
  int idx = blockIdx.x * 256 + threadIdx.x;   // k*Npad + n
  if (idx >= K * Npad) return;
  int n = idx % Npad, k = idx / Npad;
  float v = (n < N) ? W[(size_t)k * N + n] : 0.0f;
  Wt[(size_t)n * K + k] = f2bf(v);
}

// ---------------------------------------------------------------------------
// GEMM: C[M,N] fp32 = A[M,K] bf16 * Bt[N,K] bf16 (B pre-transposed).
// 256 thr = 8 waves; tile 128x128, BK=64, double-buffered async LDS staging.
// ---------------------------------------------------------------------------
#define BM 128
#define BN 128
#define BK 64

// Stage a 128x64 bf16 tile (row stride srcK elems) into LDS (row stride 64).
__device__ __forceinline__ void stage128x64(u32 ldsbase, const u16* src,
                                            int srcK, int t) {
  for (int i = 0; i < 4; ++i) {
    int idx = i * 256 + t;            // 1024 chunks of 16B
    int row = idx >> 3, c = idx & 7;  // 8 chunks per row
    async_b128(ldsbase + (u32)(idx * 16), src, (u32)((row * srcK + c * 8) * 2));
  }
}

__global__ __launch_bounds__(256) void gemm_bf16(
    const u16* __restrict__ A, const u16* __restrict__ Bt,
    float* __restrict__ C, int M, int N, int K) {
  extern __shared__ u16 gsm[];       // 2 x (As 8192 + Bs 8192) u16 = 64 KB

  const int t    = threadIdx.x;
  const int lane = t & 31;
  const int w    = t >> 5;
  const int half = lane >> 4;
  const int i16  = lane & 15;
  const int m0   = blockIdx.y * BM;
  const int n0   = blockIdx.x * BN;
  const int wm   = (w & 3) * 32;     // 4 waves tile M
  const int wn   = (w >> 2) * 64;    // 2 waves tile N

  v8f acc[2][4];
  for (int mt = 0; mt < 2; ++mt)
    for (int nt = 0; nt < 4; ++nt)
      for (int r = 0; r < 8; ++r) acc[mt][nt][r] = 0.0f;

  const u16* Arow = A  + (size_t)m0 * K;
  const u16* Brow = Bt + (size_t)n0 * K;
  const int KT = K / BK;

  stage128x64(lds_off(gsm), Arow, K, t);
  stage128x64(lds_off(gsm + 8192), Brow, K, t);

  for (int kt = 0; kt < KT; ++kt) {
    async_wait0();
    __syncthreads();
    const int cur = kt & 1;
    u16* Asc = gsm + cur * 16384;
    u16* Bsc = Asc + 8192;
    if (kt + 1 < KT) {
      u16* Asn = gsm + (cur ^ 1) * 16384;
      stage128x64(lds_off(Asn), Arow + (size_t)(kt + 1) * BK, K, t);
      stage128x64(lds_off(Asn + 8192), Brow + (size_t)(kt + 1) * BK, K, t);
    }
    for (int kk = 0; kk < 2; ++kk) {
      v16bf af0 = ldfrag(&Asc[(wm + i16) * BK + kk * 32], half);
      v16bf af1 = ldfrag(&Asc[(wm + 16 + i16) * BK + kk * 32], half);
      for (int nt = 0; nt < 4; ++nt) {
        v16bf bfr = ldfrag(&Bsc[(wn + nt * 16 + i16) * BK + kk * 32], half);
        acc[0][nt] = wmma_bf16(af0, bfr, acc[0][nt]);
        acc[1][nt] = wmma_bf16(af1, bfr, acc[1][nt]);
      }
    }
  }

  for (int mt = 0; mt < 2; ++mt)
    for (int nt = 0; nt < 4; ++nt)
      for (int r = 0; r < 8; ++r) {
        int m = m0 + wm + mt * 16 + half * 8 + r;
        int n = n0 + wn + nt * 16 + i16;
        if (n < N) C[(size_t)m * N + n] = acc[mt][nt][r];
      }
}

// ---------------------------------------------------------------------------
// Row LayerNorm: fp32 in -> bf16 out
// ---------------------------------------------------------------------------
__global__ __launch_bounds__(256) void ln_rows(
    const float* __restrict__ X, const float* __restrict__ g,
    const float* __restrict__ b, u16* __restrict__ Y,
    int len, int xstride, int ystride) {
  __shared__ float rs[256], rs2[256];
  const int row = blockIdx.x;
  const float* x = X + (size_t)row * xstride;
  u16* y = Y + (size_t)row * ystride;
  float s = 0.f, s2 = 0.f;
  for (int i = threadIdx.x; i < len; i += 256) {
    float v = x[i]; s += v; s2 += v * v;
  }
  rs[threadIdx.x] = s; rs2[threadIdx.x] = s2;
  __syncthreads();
  for (int o = 128; o > 0; o >>= 1) {
    if (threadIdx.x < o) {
      rs[threadIdx.x]  += rs[threadIdx.x + o];
      rs2[threadIdx.x] += rs2[threadIdx.x + o];
    }
    __syncthreads();
  }
  float mean = rs[0] / (float)len;
  float var  = rs2[0] / (float)len - mean * mean;
  float inv  = rsqrtf(var + 1e-6f);
  for (int i = threadIdx.x; i < len; i += 256)
    y[i] = f2bf((x[i] - mean) * inv * g[i] + b[i]);
}

// ---------------------------------------------------------------------------
// RoPE on k_pe (last 64 cols of ckv) -> kpe fp32 [B*S, 64]
// ---------------------------------------------------------------------------
__global__ __launch_bounds__(256) void rope_kpe_kernel(
    const float* __restrict__ ckv, const float* __restrict__ cosb,
    const float* __restrict__ sinb, const int* __restrict__ pos,
    float* __restrict__ kpe) {
  int idx = blockIdx.x * 256 + threadIdx.x;        // B*S*64
  int i  = idx & 63;
  int bs = idx >> 6;
  const float* x = ckv + (size_t)bs * (KVL_ + RD_) + KVL_;
  int p = pos[bs];
  float c  = cosb[p * RD_ + i];
  float sn = sinb[p * RD_ + i];
  float rot = (i < 32) ? -x[i + 32] : x[i - 32];
  kpe[idx] = x[i] * c + rot * sn;
}

// ---------------------------------------------------------------------------
// q prep: q fp32 [B,S,NH,192] -> qs bf16 [B,NH,S,192], RoPE on d>=128,
// softmax scale (192^-0.5) folded in.
// ---------------------------------------------------------------------------
__global__ __launch_bounds__(256) void q_prep_kernel(
    const float* __restrict__ q, const float* __restrict__ cosb,
    const float* __restrict__ sinb, const int* __restrict__ pos,
    u16* __restrict__ qs) {
  const float scale = 0.0721687836487032f;
  int idx = blockIdx.x * 256 + threadIdx.x;        // B*S*NH*192
  int d = idx % QHD_;
  int tmp = idx / QHD_;
  int h = tmp % NH_;
  int bs = tmp / NH_;
  int b = bs / S_, s = bs % S_;
  const float* qrow = q + (size_t)bs * (NH_ * QHD_) + (size_t)h * QHD_;
  float val;
  if (d < ND_) {
    val = qrow[d];
  } else {
    int i = d - ND_;
    int p = pos[bs];
    float c  = cosb[p * RD_ + i];
    float sn = sinb[p * RD_ + i];
    float rot = (i < 32) ? -qrow[ND_ + i + 32] : qrow[ND_ + i - 32];
    val = qrow[d] * c + rot * sn;
  }
  qs[((size_t)(b * NH_ + h) * S_ + s) * QHD_ + d] = f2bf(val * scale);
}

// ---------------------------------------------------------------------------
// kv prep: kv fp32 [B,S,NH,256] + kpe -> k bf16 [B,NH,S,192],
// vT bf16 [B,NH,VD,S] (pre-transposed for attention staging).
// ---------------------------------------------------------------------------
__global__ __launch_bounds__(256) void kv_prep_kernel(
    const float* __restrict__ kv, const float* __restrict__ kpe,
    u16* __restrict__ k, u16* __restrict__ vT) {
  int idx = blockIdx.x * 256 + threadIdx.x;        // B*S*NH*320
  int c = idx % 320;
  int tmp = idx / 320;
  int h = tmp % NH_;
  int bs = tmp / NH_;
  int b = bs / S_, s = bs % S_;
  int bh = b * NH_ + h;
  size_t src = (size_t)bs * (NH_ * (ND_ + VD_)) + (size_t)h * (ND_ + VD_);
  size_t hd  = (size_t)bh * S_ + s;
  if (c < ND_) {
    k[hd * QHD_ + c] = f2bf(kv[src + c]);
  } else if (c < ND_ + VD_) {
    int vd = c - ND_;
    vT[((size_t)bh * VD_ + vd) * S_ + s] = f2bf(kv[src + c]);
  } else {
    int i = c - (ND_ + VD_);
    k[hd * QHD_ + ND_ + i] = f2bf(kpe[(size_t)bs * RD_ + i]);
  }
}

// ---------------------------------------------------------------------------
// Flash attention (causal) with WMMA; double-buffered async K/V staging.
// grid.x = S/128 q-blocks, grid.y = B*NH. 256 threads = 8 waves,
// wave w owns q-rows [w*16, w*16+16). Dynamic LDS = 240 KB (WGP has 320 KB).
// LDS map (u16 units):
//   Qs      @ 0       (128x192 = 24576)
//   Ks[0/1] @ 24576 + cur*24576
//   Vt[0/1] @ 73728 + cur*16384   ([vd][key])
//   Ps      @ 106496  (128x128 = 16384)
// ---------------------------------------------------------------------------
#define KS_BASE_ 24576
#define VT_BASE_ 73728
#define PS_BASE_ 106496

__global__ __launch_bounds__(256) void mla_flash(
    const u16* __restrict__ qs, const u16* __restrict__ kg,
    const u16* __restrict__ vTg, u16* __restrict__ o) {
  extern __shared__ u16 fsm[];
  u16* Qs = fsm;
  u16* Ps = fsm + PS_BASE_;

  const int t    = threadIdx.x;
  const int lane = t & 31;
  const int w    = t >> 5;
  const int half = lane >> 4;
  const int i16  = lane & 15;
  const int qb   = blockIdx.x;
  const int bh   = blockIdx.y;
  const int q0   = qb * 128;
  const size_t hbase = (size_t)bh * S_;
  const u16* vhead = vTg + ((size_t)bh * VD_) * S_;

  // Issue Q tile + first K/V tile asyncs together, then one wait.
  {
    const u16* src = qs + (hbase + q0) * QHD_;
    u32 qoff = lds_off(Qs);
    for (int i = 0; i < 12; ++i) {
      int idx = i * 256 + t;
      async_b128(qoff + (u32)(idx * 16), src, (u32)(idx * 16));
    }
    const u16* ksrc = kg + hbase * QHD_;          // k0 = 0
    u32 koff = lds_off(fsm + KS_BASE_);
    for (int i = 0; i < 12; ++i) {
      int idx = i * 256 + t;
      async_b128(koff + (u32)(idx * 16), ksrc, (u32)(idx * 16));
    }
    u32 voff = lds_off(fsm + VT_BASE_);
    for (int i = 0; i < 8; ++i) {
      int idx = i * 256 + t;                      // 2048 x 16B chunks
      int vd = idx >> 4, c = idx & 15;
      async_b128(voff + (u32)(idx * 16), vhead, (u32)((vd * S_) * 2 + c * 16));
    }
  }
  async_wait0();
  __syncthreads();

  // Q fragments stay in registers for the whole block.
  v16bf qf[6];
  {
    const u16* rb = &Qs[(w * 16 + i16) * QHD_];
    for (int kk = 0; kk < 6; ++kk) qf[kk] = ldfrag(rb + kk * 32, half);
  }

  v8f oacc[8];
  float m_r[8], l_r[8];
  for (int vt = 0; vt < 8; ++vt)
    for (int r = 0; r < 8; ++r) oacc[vt][r] = 0.0f;
  for (int r = 0; r < 8; ++r) { m_r[r] = -1e30f; l_r[r] = 0.0f; }

  for (int j = 0; j <= qb; ++j) {
    const int cur = j & 1;
    // Prefetch next key-block into the other buffer while computing this one.
    if (j + 1 <= qb) {
      const int nxt = cur ^ 1;
      const int k0n = (j + 1) * 128;
      const u16* ksrc = kg + (hbase + k0n) * QHD_;
      u32 koff = lds_off(fsm + KS_BASE_ + nxt * 24576);
      for (int i = 0; i < 12; ++i) {
        int idx = i * 256 + t;
        async_b128(koff + (u32)(idx * 16), ksrc, (u32)(idx * 16));
      }
      const u16* vsrc = vhead + k0n;
      u32 voff = lds_off(fsm + VT_BASE_ + nxt * 16384);
      for (int i = 0; i < 8; ++i) {
        int idx = i * 256 + t;
        int vd = idx >> 4, c = idx & 15;
        async_b128(voff + (u32)(idx * 16), vsrc, (u32)((vd * S_) * 2 + c * 16));
      }
    }
    const u16* Ks = fsm + KS_BASE_ + cur * 24576;
    const u16* Vt = fsm + VT_BASE_ + cur * 16384;

    // S = Q * K^T  (8 n-tiles x 6 k-steps of WMMA)
    v8f sacc[8];
    for (int nt = 0; nt < 8; ++nt) {
      v8f a;
      for (int r = 0; r < 8; ++r) a[r] = 0.0f;
      const u16* rb = &Ks[(nt * 16 + i16) * QHD_];
      for (int kk = 0; kk < 6; ++kk) {
        v16bf bfr = ldfrag(rb + kk * 32, half);
        a = wmma_bf16(qf[kk], bfr, a);
      }
      sacc[nt] = a;
    }

    // Causal mask on the diagonal block (scale already folded into Q).
    if (j == qb) {
      for (int nt = 0; nt < 8; ++nt)
        for (int r = 0; r < 8; ++r) {
          int mg = w * 16 + half * 8 + r;
          int ng = nt * 16 + i16;
          if (ng > mg) sacc[nt][r] = -1e30f;
        }
    }

    // Online softmax per row (row = half*8 + r; cols in 16 lanes x 8 tiles).
    for (int r = 0; r < 8; ++r) {
      float mx = sacc[0][r];
      for (int nt = 1; nt < 8; ++nt) mx = fmaxf(mx, sacc[nt][r]);
      for (int off = 8; off > 0; off >>= 1)
        mx = fmaxf(mx, __shfl_xor(mx, off, 16));
      float mnew = fmaxf(m_r[r], mx);
      float corr = __expf(m_r[r] - mnew);
      m_r[r] = mnew;
      l_r[r] *= corr;
      for (int vt = 0; vt < 8; ++vt) oacc[vt][r] *= corr;
      float psum = 0.0f;
      int rowl = w * 16 + half * 8 + r;
      for (int nt = 0; nt < 8; ++nt) {
        float p = __expf(sacc[nt][r] - mnew);
        psum += p;
        Ps[rowl * 128 + nt * 16 + i16] = f2bf(p);
      }
      for (int off = 8; off > 0; off >>= 1)
        psum += __shfl_xor(psum, off, 16);
      l_r[r] += psum;
    }
    __syncthreads();   // Ps fully written before PV reads

    // O += P * V  (8 vd-tiles x 4 k-steps of WMMA)
    for (int vt = 0; vt < 8; ++vt) {
      const u16* pb = &Ps[(w * 16 + i16) * 128];
      const u16* vb = &Vt[(vt * 16 + i16) * 128];
      v8f a = oacc[vt];
      for (int kk = 0; kk < 4; ++kk) {
        v16bf pf = ldfrag(pb + kk * 32, half);
        v16bf vf = ldfrag(vb + kk * 32, half);
        a = wmma_bf16(pf, vf, a);
      }
      oacc[vt] = a;
    }

    // Next buffer must be resident, and all waves must be done with this
    // iteration's Ps/Ks/Vt reads, before the next iteration proceeds.
    async_wait0();
    __syncthreads();
  }

  // Epilogue: o[b, s, h*128 + vd] = bf16(O / l), feeds final GEMM directly.
  const int b = bh >> 5, h = bh & 31;
  for (int vt = 0; vt < 8; ++vt)
    for (int r = 0; r < 8; ++r) {
      int m = q0 + w * 16 + half * 8 + r;
      float val = oacc[vt][r] / l_r[r];
      o[((size_t)(b * S_ + m)) * (NH_ * VD_) + h * VD_ + vt * 16 + i16] = f2bf(val);
    }
}

// ---------------------------------------------------------------------------
// Host-side orchestration
// ---------------------------------------------------------------------------
extern "C" void kernel_launch(void* const* d_in, const int* in_sizes, int n_in,
                              void* d_out, int out_size, void* d_ws, size_t ws_size,
                              hipStream_t stream) {
  (void)in_sizes; (void)n_in; (void)out_size; (void)ws_size;

  const float* hs   = (const float*)d_in[0];
  const float* cosb = (const float*)d_in[1];
  const float* sinb = (const float*)d_in[2];
  const int*   pos  = (const int*)d_in[3];
  const float* Wqa  = (const float*)d_in[4];
  const float* gqa  = (const float*)d_in[5];
  const float* bqa  = (const float*)d_in[6];
  const float* Wqb  = (const float*)d_in[7];
  const float* Wkva = (const float*)d_in[8];
  const float* gkva = (const float*)d_in[9];
  const float* bkva = (const float*)d_in[10];
  const float* Wkvb = (const float*)d_in[11];
  const float* Wo   = (const float*)d_in[12];
  float* out = (float*)d_out;

  const int NKV  = KVL_ + RD_;        // 576
  const int NKVP = 640;               // padded to 128
  const int NQB  = NH_ * QHD_;        // 6144
  const int NKVB = NH_ * (ND_ + VD_); // 8192

  // Workspace layout
  char* p = (char*)d_ws;
  u16* hs_bf   = (u16*)p; p += (size_t)ROWS_ * H_ * 2;          // 33.6 MB
  u16* Wqa_t   = (u16*)p; p += (size_t)QL_ * H_ * 2;            // 12.6 MB
  u16* Wqb_t   = (u16*)p; p += (size_t)NQB * QL_ * 2;           // 18.9 MB
  u16* Wkva_t  = (u16*)p; p += (size_t)NKVP * H_ * 2;           // 5.3 MB
  u16* Wkvb_t  = (u16*)p; p += (size_t)NKVB * KVL_ * 2;         // 8.4 MB
  u16* Wo_t    = (u16*)p; p += (size_t)H_ * H_ * 2;             // 33.6 MB
  float* q_a   = (float*)p; p += (size_t)ROWS_ * QL_ * 4;       // 25.2 MB
  u16* q_an    = (u16*)p; p += (size_t)ROWS_ * QL_ * 2;         // 12.6 MB
  float* qbuf  = (float*)p; p += (size_t)ROWS_ * NQB * 4;       // 100.7 MB
  float* ckv   = (float*)p; p += (size_t)ROWS_ * NKV * 4;       // 9.4 MB
  u16* kv_cn   = (u16*)p; p += (size_t)ROWS_ * KVL_ * 2;        // 4.2 MB
  float* kpe   = (float*)p; p += (size_t)ROWS_ * RD_ * 4;       // 1.0 MB
  float* kvbuf = (float*)p; p += (size_t)ROWS_ * NKVB * 4;      // 134.2 MB
  u16* qs16    = (u16*)p; p += (size_t)ROWS_ * NH_ * QHD_ * 2;  // 50.3 MB
  u16* k16     = (u16*)p; p += (size_t)ROWS_ * NH_ * QHD_ * 2;  // 50.3 MB
  u16* vT16    = (u16*)p; p += (size_t)ROWS_ * NH_ * VD_ * 2;   // 33.6 MB
  u16* obuf    = (u16*)p; p += (size_t)ROWS_ * NH_ * VD_ * 2;   // 33.6 MB

  const dim3 blk(256);
  (void)hipFuncSetAttribute((const void*)gemm_bf16,
                            hipFuncAttributeMaxDynamicSharedMemorySize, 65536);
  const int flash_lds = 122880 * 2;   // 245760 B = 240 KB
  (void)hipFuncSetAttribute((const void*)mla_flash,
                            hipFuncAttributeMaxDynamicSharedMemorySize, flash_lds);

  // 0) one-time casts / weight transposes (bandwidth-trivial)
  cast_bf16_kernel<<<(ROWS_ * H_ / 4 + 255) / 256, blk, 0, stream>>>(hs, hs_bf, ROWS_ * H_ / 4);
  transpose_cast_kernel<<<((size_t)H_ * QL_ + 255) / 256, blk, 0, stream>>>(Wqa, Wqa_t, H_, QL_, QL_);
  transpose_cast_kernel<<<((size_t)QL_ * NQB + 255) / 256, blk, 0, stream>>>(Wqb, Wqb_t, QL_, NQB, NQB);
  transpose_cast_kernel<<<((size_t)H_ * NKVP + 255) / 256, blk, 0, stream>>>(Wkva, Wkva_t, H_, NKV, NKVP);
  transpose_cast_kernel<<<((size_t)KVL_ * NKVB + 255) / 256, blk, 0, stream>>>(Wkvb, Wkvb_t, KVL_, NKVB, NKVB);
  transpose_cast_kernel<<<((size_t)H_ * H_ + 255) / 256, blk, 0, stream>>>(Wo, Wo_t, H_, H_, H_);

  // 1) q_a = hs @ Wqa
  gemm_bf16<<<dim3(QL_ / BN, ROWS_ / BM), blk, 65536, stream>>>(hs_bf, Wqa_t, q_a, ROWS_, QL_, H_);
  // 2) q_an = LN(q_a) -> bf16
  ln_rows<<<ROWS_, blk, 0, stream>>>(q_a, gqa, bqa, q_an, QL_, QL_, QL_);
  // 3) q = q_an @ Wqb
  gemm_bf16<<<dim3(NQB / BN, ROWS_ / BM), blk, 65536, stream>>>(q_an, Wqb_t, qbuf, ROWS_, NQB, QL_);
  // 4) ckv = hs @ Wkva
  gemm_bf16<<<dim3(NKVP / BN, ROWS_ / BM), blk, 65536, stream>>>(hs_bf, Wkva_t, ckv, ROWS_, NKV, H_);
  // 5) kv_cn = LN(ckv[:, :512]) -> bf16
  ln_rows<<<ROWS_, blk, 0, stream>>>(ckv, gkva, bkva, kv_cn, KVL_, NKV, KVL_);
  // 6) RoPE(k_pe)
  rope_kpe_kernel<<<(ROWS_ * RD_) / 256, blk, 0, stream>>>(ckv, cosb, sinb, pos, kpe);
  // 7) kv = kv_cn @ Wkvb
  gemm_bf16<<<dim3(NKVB / BN, ROWS_ / BM), blk, 65536, stream>>>(kv_cn, Wkvb_t, kvbuf, ROWS_, NKVB, KVL_);
  // 8) qs = scale*RoPE(q) -> bf16 [B,NH,S,192]
  q_prep_kernel<<<(ROWS_ * NH_ * QHD_) / 256, blk, 0, stream>>>(qbuf, cosb, sinb, pos, qs16);
  // 9) k = [k_nope | k_pe] -> bf16 [B,NH,S,192]; vT -> bf16 [B,NH,VD,S]
  kv_prep_kernel<<<((size_t)ROWS_ * NH_ * 320) / 256, blk, 0, stream>>>(kvbuf, kpe, k16, vT16);
  // 10) flash attention -> obuf bf16 [B,S,NH*VD]
  mla_flash<<<dim3(S_ / 128, B_ * NH_), blk, flash_lds, stream>>>(qs16, k16, vT16, obuf);
  // 11) out = obuf @ Wo
  gemm_bf16<<<dim3(H_ / BN, ROWS_ / BM), blk, 65536, stream>>>(obuf, Wo_t, out, ROWS_, H_, H_);
}
